// TacticDecoder_84756884619502
// MI455X (gfx1250) — matrix operand order repaired and checked by reading, not compiled, not run on hardware
//
#include <hip/hip_runtime.h>
#include <cstdint>

// ---------------------------------------------------------------------------
// TacticDecoder step for MI455X (gfx1250, wave32, WMMA).
// All matmuls run as v_wmma_f32_16x16x32_bf16 with f32 accumulation.
// Round 3: GEMM double-buffering rewritten with named fragment registers
// (compile-time indices only) -> no scratch/LDS spills in the inner loop.
// ---------------------------------------------------------------------------

#define B_   4096
#define L_   64
#define H_   512
#define E_   256
#define S_   128
#define CT_  259
#define R_   4096
#define IN_  1670
#define INP_ 1696          // IN padded to multiple of 32
#define CTP_ 288           // CT padded to multiple of 32
#define G3H_ 1536          // 3*H

typedef __attribute__((ext_vector_type(16))) __bf16 v16bf;
typedef __attribute__((ext_vector_type(8)))  float  v8f;

union Frag16 { uint4 q[2]; v16bf v; };

__device__ __forceinline__ unsigned short f2bf(float f) {
  unsigned int u = __builtin_bit_cast(unsigned int, f);
  u += 0x7FFFu + ((u >> 16) & 1u);            // round-to-nearest-even
  return (unsigned short)(u >> 16);
}
__device__ __forceinline__ float bf2f(unsigned short h) {
  unsigned int u = ((unsigned int)h) << 16;
  return __builtin_bit_cast(float, u);
}

// ---------------------------------------------------------------------------
// Weight conversion f32 -> bf16, pre-swizzled into WMMA B-fragment order.
// B layout used here (contiguous k, as in loadA): lane = 16*((k%32)/16) + (n%16), elem e = k%16.
// dst[ ((kt*(N/16)+nt)*32 + lane)*16 + e ]
// ---------------------------------------------------------------------------
__global__ void swizzle_b_kernel(const float* __restrict__ src, int srcK, int srcLd,
                                 int N, int Kpad, int transpose,
                                 unsigned short* __restrict__ dst) {
  const size_t total = (size_t)Kpad * N;
  for (size_t i = (size_t)blockIdx.x * blockDim.x + threadIdx.x; i < total;
       i += (size_t)gridDim.x * blockDim.x) {
    int k = (int)(i / N);
    int n = (int)(i - (size_t)k * N);
    float v = 0.0f;
    if (k < srcK) v = transpose ? src[(size_t)n * srcLd + k] : src[(size_t)k * srcLd + n];
    int kt = k >> 5, nt = n >> 4;
    int lane = (((k >> 4) & 1) << 4) + (n & 15);
    int e = k & 15;
    dst[(((size_t)kt * (N >> 4) + nt) * 32 + lane) * 16 + e] = f2bf(v);
  }
}

__global__ void cvt_bf16_kernel(const float* __restrict__ src,
                                unsigned short* __restrict__ dst, size_t n) {
  for (size_t i = (size_t)blockIdx.x * blockDim.x + threadIdx.x; i < n;
       i += (size_t)gridDim.x * blockDim.x)
    dst[i] = f2bf(src[i]);
}

// Assemble GRU input x_bf [B, INP_] = [a_tm1 | p_t | n_t | ctx_env | ctx_lc | 0-pad]
// ctx regions (cols 1152..1669) are written by context_kernel.
__global__ void assemble_x_kernel(const float* __restrict__ a_tm1,
                                  const float* __restrict__ p_t,
                                  const int* __restrict__ sym_idx,
                                  const float* __restrict__ sym_table,
                                  unsigned short* __restrict__ xbf) {
  const size_t total = (size_t)B_ * INP_;
  for (size_t i = (size_t)blockIdx.x * blockDim.x + threadIdx.x; i < total;
       i += (size_t)gridDim.x * blockDim.x) {
    int b = (int)(i / INP_);
    int c = (int)(i - (size_t)b * INP_);
    float v;
    if (c < 256)        v = a_tm1[(size_t)b * E_ + c];
    else if (c < 1024)  v = p_t[(size_t)b * (H_ + E_) + (c - 256)];
    else if (c < 1152)  v = sym_table[(size_t)sym_idx[b] * S_ + (c - 1024)];
    else if (c < IN_)   continue;     // context region, filled by context_kernel
    else                v = 0.0f;     // K padding
    xbf[i] = f2bf(v);
  }
}

// ---------------------------------------------------------------------------
// Generic bf16 WMMA GEMM: C[M,N] = act(A[M,K]@Bsw + bias)
// One wave computes a 16(M) x 64(N) strip. The k-loop is manually unrolled
// by 2 with two *named* fragment buffers so everything stays in VGPRs and
// step-k WMMAs overlap step-(k+1) global loads.
// epi: 0 = store f32 (no bias), 1 = +bias store f32, 2 = +bias, tanh, store bf16
// ---------------------------------------------------------------------------
__global__ __launch_bounds__(256)
void gemm_bf16_wmma(const unsigned short* __restrict__ A, int lda,
                    const unsigned short* __restrict__ Bsw,
                    const float* __restrict__ bias,
                    void* __restrict__ Cout,
                    int M, int N, int K, int epi) {
  const int tid  = threadIdx.x;
  const int lane = tid & 31;
  const int waveId = blockIdx.x * 8 + (tid >> 5);
  const int TM = M >> 4, TN64 = N >> 6;
  if (waveId >= TM * TN64) return;
  const int mt = waveId % TM, nt64 = waveId / TM;
  const int m0 = mt << 4, n0 = nt64 << 6;
  const int tn16 = N >> 4;
  const int lhalf = lane >> 4, l16 = lane & 15;
  const int nt16_0 = n0 >> 4;

  v8f acc[4] = {v8f{}, v8f{}, v8f{}, v8f{}};
  const unsigned short* arow = A + (size_t)(m0 + l16) * lda;

  auto loadA = [&](int kt, Frag16& f) {
    const uint4* ap = (const uint4*)(arow + kt * 32 + 16 * lhalf);
    f.q[0] = ap[0];
    f.q[1] = ap[1];                   // +8 bf16 = +1 uint4
  };
  auto loadB = [&](int kt, Frag16& f0, Frag16& f1, Frag16& f2, Frag16& f3) {
    const unsigned short* bb =
        Bsw + (((size_t)kt * tn16 + nt16_0) * 32 + lane) * 16;
    __builtin_prefetch(bb + 2 * (size_t)tn16 * 512, 0, 1);  // k+2 slab
    const uint4* bp = (const uint4*)bb;
    f0.q[0] = bp[0];   f0.q[1] = bp[1];
    f1.q[0] = bp[64];  f1.q[1] = bp[65];    // +512 ushort = +64 uint4
    f2.q[0] = bp[128]; f2.q[1] = bp[129];
    f3.q[0] = bp[192]; f3.q[1] = bp[193];
  };

#define WMMA4(FA, FB0, FB1, FB2, FB3)                                          \
  acc[0] = __builtin_amdgcn_wmma_f32_16x16x32_bf16(                            \
      false, (FA).v, false, (FB0).v, (short)0, acc[0], false, false);          \
  acc[1] = __builtin_amdgcn_wmma_f32_16x16x32_bf16(                            \
      false, (FA).v, false, (FB1).v, (short)0, acc[1], false, false);          \
  acc[2] = __builtin_amdgcn_wmma_f32_16x16x32_bf16(                            \
      false, (FA).v, false, (FB2).v, (short)0, acc[2], false, false);          \
  acc[3] = __builtin_amdgcn_wmma_f32_16x16x32_bf16(                            \
      false, (FA).v, false, (FB3).v, (short)0, acc[3], false, false)

  const int KT = K >> 5;
  Frag16 a0, a1, b00, b01, b02, b03, b10, b11, b12, b13;
  loadA(0, a0);
  loadB(0, b00, b01, b02, b03);

  int kt = 0;
  for (; kt + 1 < KT; kt += 2) {
    loadA(kt + 1, a1);
    loadB(kt + 1, b10, b11, b12, b13);
    WMMA4(a0, b00, b01, b02, b03);            // overlaps buf1 loads
    if (kt + 2 < KT) {
      loadA(kt + 2, a0);
      loadB(kt + 2, b00, b01, b02, b03);
    }
    WMMA4(a1, b10, b11, b12, b13);            // overlaps buf0 loads
  }
  if (KT & 1) {                               // odd tail (e.g. K=1696, KT=53)
    WMMA4(a0, b00, b01, b02, b03);
  }
#undef WMMA4

#pragma unroll
  for (int j = 0; j < 4; ++j) {
    const int n = n0 + j * 16 + l16;
    const float bv = (epi != 0 && bias) ? bias[n] : 0.0f;
#pragma unroll
    for (int r = 0; r < 8; ++r) {
      const int m = m0 + r + 8 * lhalf;
      float v = acc[j][r] + bv;
      if (epi == 2) {
        ((unsigned short*)Cout)[(size_t)m * N + n] = f2bf(tanhf(v));
      } else {
        ((float*)Cout)[(size_t)m * N + n] = v;
      }
    }
  }
}

// ---------------------------------------------------------------------------
// Context read: one block per (b, which). emb tile staged once in LDS (bf16),
// reused for WMMA A-fragments and for the softmax-weighted context sum.
// kt-outer loop with all 16 accumulators live: each W1e fragment is read from
// L2 exactly once per block. Fully unrolled -> fragments stay in VGPRs.
// hidden = relu(state_pre[b] + emb @ W1e) ; logit = hidden . W2 ; softmax ; ctx.
// ---------------------------------------------------------------------------
__global__ __launch_bounds__(256)
void context_kernel(const float* __restrict__ env_emb, const int* __restrict__ env_len,
                    const float* __restrict__ lc_emb, const int* __restrict__ lc_len,
                    const float* __restrict__ state_pre,      // [B,H] = s@W1s + b1
                    const unsigned short* __restrict__ W1e_sw, // swizzled [CTP_, H]
                    const float* __restrict__ W2,              // [H]
                    unsigned short* __restrict__ xbf) {        // [B, INP_]
  const int b = blockIdx.x;
  const int which = blockIdx.y;
  const float* emb = (which == 0 ? env_emb : lc_emb) + (size_t)b * (L_ * CT_);
  const int len = (which == 0 ? env_len : lc_len)[b];

  __shared__ unsigned short Abf[L_ * CTP_];  // 36.9 KB
  __shared__ float w2s[H_];
  __shared__ float logit[L_];
  __shared__ float wsm[L_];

  const int tid = threadIdx.x;
  for (int i = tid; i < L_ * CTP_; i += 256) {
    int l = i / CTP_, k = i - l * CTP_;
    Abf[i] = f2bf(k < CT_ ? emb[(size_t)l * CT_ + k] : 0.0f);
  }
  for (int i = tid; i < H_; i += 256) w2s[i] = W2[i];
  if (tid < L_) logit[tid] = 0.0f;
  __syncthreads();

  const int lane = tid & 31;
  const int wv = tid >> 5;                 // wave -> 64-column strip
  const int lhalf = lane >> 4, l16 = lane & 15;
  const int n0 = wv * 64;
  const float* sp = state_pre + (size_t)b * H_;

  v8f acc[4][4];                           // [mt][j]
#pragma unroll
  for (int mt = 0; mt < 4; ++mt)
#pragma unroll
    for (int j = 0; j < 4; ++j) acc[mt][j] = v8f{};

#pragma unroll
  for (int kt = 0; kt < CTP_ / 32; ++kt) {
    const unsigned short* bb =
        W1e_sw + (((size_t)kt * (H_ >> 4) + (n0 >> 4)) * 32 + lane) * 16;
    __builtin_prefetch(bb + (size_t)(H_ >> 4) * 512, 0, 1);
    Frag16 fb[4];
#pragma unroll
    for (int j = 0; j < 4; ++j) {
      const uint4* bp = (const uint4*)(bb + j * 512);
      fb[j].q[0] = bp[0];
      fb[j].q[1] = bp[1];
    }
#pragma unroll
    for (int mt = 0; mt < 4; ++mt) {
      Frag16 fa;
      const uint4* ap =
          (const uint4*)(Abf + (mt * 16 + l16) * CTP_ + kt * 32 + 16 * lhalf);
      fa.q[0] = ap[0];
      fa.q[1] = ap[1];
#pragma unroll
      for (int j = 0; j < 4; ++j)
        acc[mt][j] = __builtin_amdgcn_wmma_f32_16x16x32_bf16(
            false, fa.v, false, fb[j].v, (short)0, acc[mt][j], false, false);
    }
  }

  for (int mt = 0; mt < 4; ++mt) {
    float s[8] = {0, 0, 0, 0, 0, 0, 0, 0};
#pragma unroll
    for (int j = 0; j < 4; ++j) {
      const int n = n0 + j * 16 + l16;
      const float w2v = w2s[n], spv = sp[n];
#pragma unroll
      for (int r = 0; r < 8; ++r) {
        float t = acc[mt][j][r] + spv;
        t = t > 0.0f ? t : 0.0f;           // relu
        s[r] += t * w2v;
      }
    }
#pragma unroll
    for (int r = 0; r < 8; ++r)
      atomicAdd(&logit[mt * 16 + r + 8 * lhalf], s[r]);  // ds_add_f32
  }
  __syncthreads();

  if (tid == 0) {
    if (len <= 0) {
      for (int l = 0; l < L_; ++l) wsm[l] = 0.0f;        // default zero context
    } else {
      float mx = -3.0e38f;
      for (int l = 0; l < L_; ++l) {
        float v = (l < len) ? logit[l] : -1.0e9f;
        logit[l] = v;
        mx = v > mx ? v : mx;
      }
      float ssum = 0.0f;
      for (int l = 0; l < L_; ++l) {
        float e = __expf(logit[l] - mx);
        wsm[l] = e;
        ssum += e;
      }
      float inv = 1.0f / ssum;
      for (int l = 0; l < L_; ++l) wsm[l] *= inv;
    }
  }
  __syncthreads();

  unsigned short* dst = xbf + (size_t)b * INP_ + (which == 0 ? 1152 : 1411);
  for (int c = tid; c < CT_; c += 256) {
    float s2 = 0.0f;
    for (int l = 0; l < L_; ++l) s2 += wsm[l] * bf2f(Abf[l * CTP_ + c]);
    dst[c] = f2bf(s2);
  }
}

// ---------------------------------------------------------------------------
// Fused GRU gates: r,z,n -> h_new (bf16 for the dec GEMM).
// gx/gh already include b_ih/b_hh from the GEMM epilogue.
// ---------------------------------------------------------------------------
__global__ void gru_gates_kernel(const float* __restrict__ gx,
                                 const float* __restrict__ gh,
                                 const float* __restrict__ s_tm1,
                                 unsigned short* __restrict__ hbf) {
  const size_t total = (size_t)B_ * H_;
  for (size_t i = (size_t)blockIdx.x * blockDim.x + threadIdx.x; i < total;
       i += (size_t)gridDim.x * blockDim.x) {
    int b = (int)(i >> 9), n = (int)(i & (H_ - 1));
    const float* gxr = gx + (size_t)b * G3H_;
    const float* ghr = gh + (size_t)b * G3H_;
    float r = 1.0f / (1.0f + __expf(-(gxr[n] + ghr[n])));
    float z = 1.0f / (1.0f + __expf(-(gxr[n + H_] + ghr[n + H_])));
    float nn = tanhf(gxr[n + 2 * H_] + r * ghr[n + 2 * H_]);
    float h = (1.0f - z) * nn + z * s_tm1[i];
    hbf[i] = f2bf(h);
  }
}

// ---------------------------------------------------------------------------
extern "C" void kernel_launch(void* const* d_in, const int* in_sizes, int n_in,
                              void* d_out, int out_size, void* d_ws, size_t ws_size,
                              hipStream_t stream) {
  (void)in_sizes; (void)n_in; (void)out_size; (void)ws_size;
  const float* s_tm1  = (const float*)d_in[0];
  const float* a_tm1  = (const float*)d_in[1];
  const float* p_t    = (const float*)d_in[2];
  const int*   symIdx = (const int*)d_in[3];
  const float* env_e  = (const float*)d_in[4];
  const int*   env_l  = (const int*)d_in[5];
  const float* lc_e   = (const float*)d_in[6];
  const int*   lc_l   = (const int*)d_in[7];
  const float* symTab = (const float*)d_in[8];
  const float* W1     = (const float*)d_in[9];
  const float* b1     = (const float*)d_in[10];
  const float* W2     = (const float*)d_in[11];
  // d_in[12] = b2 (scalar): uniform logit shift -> softmax-invariant, skipped.
  const float* W_ih   = (const float*)d_in[13];
  const float* W_hh   = (const float*)d_in[14];
  const float* b_ih   = (const float*)d_in[15];
  const float* b_hh   = (const float*)d_in[16];
  const float* Wd     = (const float*)d_in[17];
  const float* bd     = (const float*)d_in[18];
  const float* rule   = (const float*)d_in[19];

  char* ws = (char*)d_ws;
  size_t off = 0;
  auto alloc = [&](size_t bytes) -> char* {
    char* p = ws + off;
    off = (off + bytes + 255) & ~(size_t)255;
    return p;
  };
  unsigned short* s_bf     = (unsigned short*)alloc((size_t)B_ * H_ * 2);
  unsigned short* x_bf     = (unsigned short*)alloc((size_t)B_ * INP_ * 2);
  unsigned short* W1s_sw   = (unsigned short*)alloc((size_t)H_ * H_ * 2);
  unsigned short* W1e_sw   = (unsigned short*)alloc((size_t)CTP_ * H_ * 2);
  unsigned short* Wih_sw   = (unsigned short*)alloc((size_t)INP_ * G3H_ * 2);
  unsigned short* Whh_sw   = (unsigned short*)alloc((size_t)H_ * G3H_ * 2);
  unsigned short* Wd_sw    = (unsigned short*)alloc((size_t)H_ * E_ * 2);
  unsigned short* ruleT_sw = (unsigned short*)alloc((size_t)E_ * R_ * 2);
  float*          statePre = (float*)alloc((size_t)B_ * H_ * 4);
  float*          gx       = (float*)alloc((size_t)B_ * G3H_ * 4);
  float*          gh       = (float*)alloc((size_t)B_ * G3H_ * 4);
  unsigned short* h_bf     = (unsigned short*)alloc((size_t)B_ * H_ * 2);
  unsigned short* dec_bf   = (unsigned short*)alloc((size_t)B_ * E_ * 2);

  auto grid1d = [](size_t n) -> int {
    size_t g = (n + 255) / 256;
    return (int)(g > 8192 ? 8192 : g);
  };

  // --- weight prep (bf16 + B-fragment swizzle) -----------------------------
  swizzle_b_kernel<<<grid1d((size_t)H_ * H_), 256, 0, stream>>>(
      W1, H_, H_, H_, H_, 0, W1s_sw);                       // W1[:512]
  swizzle_b_kernel<<<grid1d((size_t)CTP_ * H_), 256, 0, stream>>>(
      W1 + (size_t)H_ * H_, CT_, H_, H_, CTP_, 0, W1e_sw);  // W1[512:771]
  swizzle_b_kernel<<<grid1d((size_t)INP_ * G3H_), 256, 0, stream>>>(
      W_ih, IN_, G3H_, G3H_, INP_, 0, Wih_sw);
  swizzle_b_kernel<<<grid1d((size_t)H_ * G3H_), 256, 0, stream>>>(
      W_hh, H_, G3H_, G3H_, H_, 0, Whh_sw);
  swizzle_b_kernel<<<grid1d((size_t)H_ * E_), 256, 0, stream>>>(
      Wd, H_, E_, E_, H_, 0, Wd_sw);
  swizzle_b_kernel<<<grid1d((size_t)E_ * R_), 256, 0, stream>>>(
      rule, E_, E_, R_, E_, 1, ruleT_sw);                   // rule_table^T

  cvt_bf16_kernel<<<grid1d((size_t)B_ * H_), 256, 0, stream>>>(
      s_tm1, s_bf, (size_t)B_ * H_);
  assemble_x_kernel<<<grid1d((size_t)B_ * INP_), 256, 0, stream>>>(
      a_tm1, p_t, symIdx, symTab, x_bf);

  // --- state_pre = s_tm1 @ W1[:H] + b1  (M=4096,N=512,K=512) ---------------
  gemm_bf16_wmma<<<(B_ / 16) * (H_ / 64) / 8, 256, 0, stream>>>(
      s_bf, H_, W1s_sw, b1, statePre, B_, H_, H_, 1);

  // --- context reads (env + lc), write ctx straight into x_bf --------------
  context_kernel<<<dim3(B_, 2), 256, 0, stream>>>(
      env_e, env_l, lc_e, lc_l, statePre, W1e_sw, W2, x_bf);

  // --- GRU: gx = x@W_ih + b_ih ; gh = s@W_hh + b_hh ------------------------
  gemm_bf16_wmma<<<(B_ / 16) * (G3H_ / 64) / 8, 256, 0, stream>>>(
      x_bf, INP_, Wih_sw, b_ih, gx, B_, G3H_, INP_, 1);
  gemm_bf16_wmma<<<(B_ / 16) * (G3H_ / 64) / 8, 256, 0, stream>>>(
      s_bf, H_, Whh_sw, b_hh, gh, B_, G3H_, H_, 1);
  gru_gates_kernel<<<grid1d((size_t)B_ * H_), 256, 0, stream>>>(
      gx, gh, s_tm1, h_bf);

  // --- dec = tanh(h_new @ Wd + bd) ; logits = dec @ rule_table^T -----------
  gemm_bf16_wmma<<<(B_ / 16) * (E_ / 64) / 8, 256, 0, stream>>>(
      h_bf, H_, Wd_sw, bd, dec_bf, B_, E_, H_, 2);
  gemm_bf16_wmma<<<(B_ / 16) * (R_ / 64) / 8, 256, 0, stream>>>(
      dec_bf, E_, ruleT_sw, nullptr, d_out, B_, R_, E_, 0);
}
